// CurlyWrapperWithMetricsCFD_57604101374184
// MI455X (gfx1250) — matrix operand order repaired and checked by reading, not compiled, not run on hardware
//
#include <hip/hip_runtime.h>
#include <hip/hip_bf16.h>
#include <math.h>

typedef _Float16 half_t;
typedef __attribute__((ext_vector_type(8)))  _Float16 v8h;
typedef __attribute__((ext_vector_type(16))) _Float16 v16h;
typedef __attribute__((ext_vector_type(8)))  float    v8f;

#define HDIM 512
#define KNN_THREADS 512
#define QPB 8            // queries per block in KNN kernel

// ---------------- Layer 1: h1 = relu([x, t] @ W1 + b1), stored f16 ----------------
__global__ void k_layer1(const float* __restrict__ z, const float* __restrict__ t,
                         const float* __restrict__ W1, const float* __restrict__ b1,
                         half_t* __restrict__ h1, int B) {
  int idx = blockIdx.x * blockDim.x + threadIdx.x;
  if (idx >= B * HDIM) return;
  int b = idx >> 9;          // / HDIM
  int j = idx & (HDIM - 1);
  const float* zr = z + b * 6;
  float tv = t[0];
  float s = zr[0] * W1[0 * HDIM + j] + zr[1] * W1[1 * HDIM + j] +
            zr[2] * W1[2 * HDIM + j] + tv * W1[3 * HDIM + j] + b1[j];
  s = s > 0.f ? s : 0.f;
  h1[idx] = (half_t)s;
}

// ------------- W2 transpose + f16 convert: w2t[n][k] = (f16) W2[k][n] -------------
__global__ void k_w2t(const float* __restrict__ W2, half_t* __restrict__ w2t) {
  int idx = blockIdx.x * blockDim.x + threadIdx.x;   // HDIM*HDIM threads
  int n = idx >> 9;
  int k = idx & (HDIM - 1);
  w2t[n * HDIM + k] = (half_t)W2[k * HDIM + n];
}

// ------------- Layer 2 GEMM via WMMA: h2 = relu(h1 @ W2 + b2), f32 out -------------
// One wave computes one 16x16 tile of C. A = h1 [B][H] f16 row-major,
// BT = W2^T [H][H] f16 (so B-fragment column loads are K-contiguous).
__global__ void k_gemm_wmma(const half_t* __restrict__ A,
                            const half_t* __restrict__ BT,
                            const float*  __restrict__ b2,
                            float* __restrict__ C) {
  int wave  = blockIdx.x * (blockDim.x >> 5) + (threadIdx.x >> 5);
  int lane  = threadIdx.x & 31;
  const int nTiles = HDIM / 16;            // 32
  int mTile = wave / nTiles;
  int nTile = wave % nTiles;
  int hi = lane >> 4;                      // half-wave select
  int lm = lane & 15;
  int mRow = mTile * 16 + lm;              // A row owned by this lane
  int nCol = nTile * 16 + lm;              // B column owned by this lane

  const half_t* arow = A  + (size_t)mRow * HDIM;
  const half_t* brow = BT + (size_t)nCol * HDIM;

  v8f c = {};
  #pragma unroll 4
  for (int k0 = 0; k0 < HDIM; k0 += 32) {
    // A 16x32 f16 fragment: elems 0..7 = K[k0+8*hi .. +7], 8..15 = K[k0+16+8*hi .. +7]
    v8h alo = *(const v8h*)(arow + k0 + 8 * hi);
    v8h ahi = *(const v8h*)(arow + k0 + 16 + 8 * hi);
    // B 32x16 f16 fragment: elems 0..7 = K[k0+8*hi .. +7], 8..15 = K[k0+16+8*hi .. +7] of column nCol
    v8h blo = *(const v8h*)(brow + k0 + 8 * hi);
    v8h bhi = *(const v8h*)(brow + k0 + 16 + 8 * hi);
    v16h a  = __builtin_shufflevector(alo, ahi, 0,1,2,3,4,5,6,7,8,9,10,11,12,13,14,15);
    v16h bb = __builtin_shufflevector(blo, bhi, 0,1,2,3,4,5,6,7,8,9,10,11,12,13,14,15);
    c = __builtin_amdgcn_wmma_f32_16x16x32_f16(false, a, false, bb,
                                               (short)0, c, false, false);
  }

  // D layout: VGPR j -> M = mTile*16 + 8*hi + j, N = nCol
  float bias = b2[nCol];
  int mBase = mTile * 16 + 8 * hi;
  #pragma unroll
  for (int j = 0; j < 8; ++j) {
    float v = c[j] + bias;
    v = v > 0.f ? v : 0.f;
    C[(size_t)(mBase + j) * HDIM + nCol] = v;
  }
}

// ------------- KNN Gaussian: QPB queries/block, exact top-k radix select -------------
// Each thread loads a point once and serves all QPB queries from registers:
// L2 traffic drops from B*5*1.2MB (12.3 GB) to (B/QPB)*5*1.2MB (~1.5 GB).
__global__ void __launch_bounds__(KNN_THREADS)
k_knn(const float* __restrict__ z,
      const float* __restrict__ x0, const float* __restrict__ x1,
      const float* __restrict__ v0, const float* __restrict__ v1,
      const int* __restrict__ kptr,
      float* __restrict__ ut, int N) {
  __shared__ unsigned int hist[QPB][256];
  __shared__ unsigned int s_prefix[QPB];
  __shared__ int s_kk[QPB];
  __shared__ float rbuf[(KNN_THREADS / 32) * 4];

  const int qbase = blockIdx.x * QPB;
  const int tid   = threadIdx.x;
  const int lane  = tid & 31;
  const int wid   = tid >> 5;

  // Query coordinates (uniform per block -> scalar loads)
  float qx[QPB], qy[QPB], qz[QPB];
  #pragma unroll
  for (int q = 0; q < QPB; ++q) {
    const float* zr = z + (size_t)(qbase + q) * 6;
    qx[q] = zr[0]; qy[q] = zr[1]; qz[q] = zr[2];
  }

  if (tid < QPB) { s_kk[tid] = kptr[0]; s_prefix[tid] = 0u; }
  __syncthreads();

  // ---- 4 radix rounds (8 bits each, MSB first) over d2 float bits ----
  for (int round = 0; round < 4; ++round) {
    unsigned pref[QPB];
    #pragma unroll
    for (int q = 0; q < QPB; ++q) pref[q] = s_prefix[q];

    for (int i = tid; i < QPB * 256; i += KNN_THREADS) ((unsigned*)hist)[i] = 0u;
    __syncthreads();

    const int shift = 24 - 8 * round;
    for (int seg = 0; seg < 2; ++seg) {
      const float* P = seg ? x1 : x0;
      for (int i = tid; i < N; i += KNN_THREADS) {
        __builtin_prefetch(&P[3 * (i + KNN_THREADS)], 0, 1);
        float px = P[3 * i], py = P[3 * i + 1], pz = P[3 * i + 2];
        #pragma unroll
        for (int q = 0; q < QPB; ++q) {
          float dx = qx[q] - px, dy = qy[q] - py, dz = qz[q] - pz;
          float d2 = dx * dx + dy * dy + dz * dz;   // >=0 -> bits order like uint
          unsigned u = __float_as_uint(d2);
          bool ok = (round == 0) || ((u >> (shift + 8)) == pref[q]);
          if (ok) atomicAdd(&hist[q][(u >> shift) & 255u], 1u);
        }
      }
    }
    __syncthreads();

    if (tid < QPB) {
      int kk = s_kk[tid];
      unsigned cum = 0, before = 0; int sel = 255;
      for (int j = 0; j < 256; ++j) {
        unsigned c = hist[tid][j];
        if (cum + c >= (unsigned)kk) { sel = j; before = cum; break; }
        cum += c;
      }
      s_prefix[tid] = (s_prefix[tid] << 8) | (unsigned)sel;
      s_kk[tid] = kk - (int)before;
    }
    __syncthreads();
  }

  // ---- thresholds per query ----
  unsigned ukth[QPB]; float inv2h2[QPB];
  #pragma unroll
  for (int q = 0; q < QPB; ++q) {
    ukth[q] = s_prefix[q];                       // bits of k-th smallest d2
    float hthr = fmaxf(sqrtf(__uint_as_float(ukth[q])), 1e-12f);  // EPS_KNN floor
    inv2h2[q] = 1.0f / (2.0f * hthr * hthr);
  }

  // ---- weighted accumulation pass ----
  float sw[QPB], a0[QPB], a1[QPB], a2[QPB];
  #pragma unroll
  for (int q = 0; q < QPB; ++q) { sw[q] = 0.f; a0[q] = 0.f; a1[q] = 0.f; a2[q] = 0.f; }

  for (int seg = 0; seg < 2; ++seg) {
    const float* P = seg ? x1 : x0;
    const float* V = seg ? v1 : v0;
    for (int i = tid; i < N; i += KNN_THREADS) {
      float px = P[3 * i], py = P[3 * i + 1], pz = P[3 * i + 2];
      float vx = V[3 * i], vy = V[3 * i + 1], vz = V[3 * i + 2];
      #pragma unroll
      for (int q = 0; q < QPB; ++q) {
        float dx = qx[q] - px, dy = qy[q] - py, dz = qz[q] - pz;
        float d2 = dx * dx + dy * dy + dz * dz;
        if (__float_as_uint(d2) <= ukth[q]) {
          float w = expf(-d2 * inv2h2[q]);
          sw[q] += w;
          a0[q] += w * vx; a1[q] += w * vy; a2[q] += w * vz;
        }
      }
    }
  }

  // ---- deterministic reduction: shuffle within wave, then across waves ----
  for (int q = 0; q < QPB; ++q) {
    float w0 = sw[q], w1 = a0[q], w2 = a1[q], w3 = a2[q];
    #pragma unroll
    for (int off = 16; off > 0; off >>= 1) {
      w0 += __shfl_down(w0, off);
      w1 += __shfl_down(w1, off);
      w2 += __shfl_down(w2, off);
      w3 += __shfl_down(w3, off);
    }
    if (lane == 0) {
      rbuf[wid * 4 + 0] = w0; rbuf[wid * 4 + 1] = w1;
      rbuf[wid * 4 + 2] = w2; rbuf[wid * 4 + 3] = w3;
    }
    __syncthreads();
    if (tid == 0) {
      float t0 = 0.f, t1 = 0.f, t2 = 0.f, t3 = 0.f;
      for (int w = 0; w < KNN_THREADS / 32; ++w) {
        t0 += rbuf[w * 4 + 0]; t1 += rbuf[w * 4 + 1];
        t2 += rbuf[w * 4 + 2]; t3 += rbuf[w * 4 + 3];
      }
      float inv = 1.0f / (t0 + 1e-12f);          // normalize by (sum w + EPS_KNN)
      float* o = ut + (size_t)(qbase + q) * 3;
      o[0] = t1 * inv; o[1] = t2 * inv; o[2] = t3 * inv;
    }
    __syncthreads();
  }
}

// ------------- Layer 3 + metrics: one wave per output row -------------
__global__ void k_out(const float* __restrict__ h2, const float* __restrict__ W3,
                      const float* __restrict__ b3, const float* __restrict__ ut,
                      float* __restrict__ out, int B) {
  int b    = blockIdx.x * (blockDim.x >> 5) + (threadIdx.x >> 5);
  int lane = threadIdx.x & 31;
  if (b >= B) return;
  float a0 = 0.f, a1 = 0.f, a2 = 0.f;
  const float* hr = h2 + (size_t)b * HDIM;
  for (int k = lane; k < HDIM; k += 32) {
    float hv = hr[k];
    a0 += hv * W3[k * 3 + 0];
    a1 += hv * W3[k * 3 + 1];
    a2 += hv * W3[k * 3 + 2];
  }
  #pragma unroll
  for (int off = 16; off > 0; off >>= 1) {
    a0 += __shfl_down(a0, off);
    a1 += __shfl_down(a1, off);
    a2 += __shfl_down(a2, off);
  }
  if (lane == 0) {
    float xd0 = a0 + b3[0], xd1 = a1 + b3[1], xd2 = a2 + b3[2];
    float u0 = ut[b * 3 + 0], u1 = ut[b * 3 + 1], u2 = ut[b * 3 + 2];
    float nu = fmaxf(sqrtf(u0 * u0 + u1 * u1 + u2 * u2), 1e-8f);
    float nd = fmaxf(sqrtf(xd0 * xd0 + xd1 * xd1 + xd2 * xd2), 1e-8f);
    float dotp = u0 * xd0 + u1 * xd1 + u2 * xd2;
    float cosd = 1.0f - dotp / (nu * nd);
    float e0 = u0 - xd0, e1 = u1 - xd1, e2 = u2 - xd2;
    float l2 = e0 * e0 + e1 * e1 + e2 * e2;
    float* o = out + (size_t)b * 6;
    o[0] = xd0; o[1] = xd1; o[2] = xd2; o[3] = cosd; o[4] = cosd; o[5] = l2;
  }
}

extern "C" void kernel_launch(void* const* d_in, const int* in_sizes, int n_in,
                              void* d_out, int out_size, void* d_ws, size_t ws_size,
                              hipStream_t stream) {
  const float* z  = (const float*)d_in[0];
  const float* t  = (const float*)d_in[1];
  const float* x0 = (const float*)d_in[2];
  const float* x1 = (const float*)d_in[3];
  const float* v0 = (const float*)d_in[4];
  const float* v1 = (const float*)d_in[5];
  const float* W1 = (const float*)d_in[6];
  const float* b1 = (const float*)d_in[7];
  const float* W2 = (const float*)d_in[8];
  const float* b2 = (const float*)d_in[9];
  const float* W3 = (const float*)d_in[10];
  const float* b3 = (const float*)d_in[11];
  const int*   kp = (const int*)d_in[12];
  float* out = (float*)d_out;

  int B = in_sizes[0] / 6;    // 2048
  int N = in_sizes[2] / 3;    // 50000

  char* ws = (char*)d_ws;
  size_t off = 0;
  half_t* h1  = (half_t*)(ws + off); off += (size_t)B * HDIM * sizeof(half_t);
  half_t* w2t = (half_t*)(ws + off); off += (size_t)HDIM * HDIM * sizeof(half_t);
  float*  h2  = (float*)(ws + off);  off += (size_t)B * HDIM * sizeof(float);
  float*  ut  = (float*)(ws + off);  off += (size_t)B * 3 * sizeof(float);

  k_layer1<<<(B * HDIM + 255) / 256, 256, 0, stream>>>(z, t, W1, b1, h1, B);
  k_w2t<<<(HDIM * HDIM + 255) / 256, 256, 0, stream>>>(W2, w2t);

  int totalWaves = (B / 16) * (HDIM / 16);   // 4096 tiles
  k_gemm_wmma<<<totalWaves / 4, 128, 0, stream>>>(h1, w2t, b2, h2);

  k_knn<<<B / QPB, KNN_THREADS, 0, stream>>>(z, x0, x1, v0, v1, kp, ut, N);

  k_out<<<(B + 7) / 8, 256, 0, stream>>>(h2, W3, b3, ut, out, B);
}
